// SimpleAttentionWithHilbert_69776038690875
// MI455X (gfx1250) — compile-verified, matrix operand order checked by reading, not run on hardware
//
#include <hip/hip_runtime.h>

// ---------------------------------------------------------------------------
// SimpleAttentionWithHilbert for MI455X (gfx1250, wave32, WMMA)
//   DIM=512, HEADS=8, HEAD_DIM=64, SEG=128, B=2, N=8192, S=64
// All matmuls: v_wmma_f32_16x16x32_bf16 (fp32 accumulate).
// All fp32->bf16 conversion hoisted out of GEMM inner loops; WMMA operands
// are fed by aligned 16/32-byte loads (global b128 / ds_load_b128).
// ---------------------------------------------------------------------------

typedef __attribute__((ext_vector_type(16))) __bf16 v16bf;
typedef __attribute__((ext_vector_type(8)))  __bf16 v8bf;
typedef __attribute__((ext_vector_type(8)))  float  v8f;

#define NB 2
#define NTOK 8192
#define CDIM 512
#define NH 8
#define HD 64
#define SEGL 128
#define NSEG 64
#define C3 1536

__device__ __forceinline__ __bf16 f2bf(float f) {
    unsigned u = __builtin_bit_cast(unsigned, f);
    unsigned r = (u + 0x7FFFu + ((u >> 16) & 1u)) >> 16;
    unsigned short s = (unsigned short)r;
    return __builtin_bit_cast(__bf16, s);
}
__device__ __forceinline__ unsigned short f2us(float f) {
    unsigned u = __builtin_bit_cast(unsigned, f);
    return (unsigned short)((u + 0x7FFFu + ((u >> 16) & 1u)) >> 16);
}
// 16-bit WMMA operand layout: element e of the v16bf holds K = kmap(e, lane>=16)
__device__ __forceinline__ int kmap(int e, int kgrp) {
    return (e & 7) + 8 * kgrp + 16 * (e >> 3);
}
__device__ __forceinline__ v16bf cat8(v8bf lo, v8bf hi) {
    return __builtin_shufflevector(lo, hi, 0, 1, 2, 3, 4, 5, 6, 7,
                                   8, 9, 10, 11, 12, 13, 14, 15);
}
// per-lane A/B operand from a row-major bf16 row: two contiguous 8-elem runs
__device__ __forceinline__ v16bf load_op(const unsigned short* row, int k0, int kgrp) {
    v8bf lo = *(const v8bf*)(row + k0 + 8 * kgrp);
    v8bf hi = *(const v8bf*)(row + k0 + 16 + 8 * kgrp);
    return cat8(lo, hi);
}

// ---------------------------------------------------------------------------
// Kernel 1: Hilbert distance on a 128x128 grid (first 8192 row-major cells)
// ---------------------------------------------------------------------------
__global__ void hilbert_d_kernel(int* __restrict__ dvals) {
    int idx = blockIdx.x * blockDim.x + threadIdx.x;
    if (idx >= NTOK) return;
    int x = idx & 127, y = idx >> 7;
    int d = 0;
    for (int s = 64; s > 0; s >>= 1) {
        int rx = (x & s) ? 1 : 0;
        int ry = (y & s) ? 1 : 0;
        d += s * s * ((3 * rx) ^ ry);
        if (ry == 0) {
            if (rx == 1) { x = s - 1 - x; y = s - 1 - y; }
            int t = x; x = y; y = t;
        }
    }
    dvals[idx] = d;
}

// ---------------------------------------------------------------------------
// Kernel 2: rank-by-count (Hilbert distances are distinct) -> perm
// ---------------------------------------------------------------------------
__global__ void hilbert_rank_kernel(const int* __restrict__ dvals,
                                    int* __restrict__ perm) {
    __shared__ int sd[256];
    int idx = blockIdx.x * 256 + threadIdx.x;
    int di = dvals[idx];
    int cnt = 0;
    for (int base = 0; base < NTOK; base += 256) {
        sd[threadIdx.x] = dvals[base + threadIdx.x];
        __syncthreads();
        #pragma unroll 8
        for (int j = 0; j < 256; ++j) cnt += (sd[j] < di) ? 1 : 0;
        __syncthreads();
    }
    perm[cnt] = idx;
}

// ---------------------------------------------------------------------------
// Kernel 3: swizzle a fp32 weight matrix (K x Ncols, row-major) into the
// per-lane bf16 WMMA B-operand layout:
//   Bsw[((tn*KSTEPS + kstep)*32 + lane)*16 + e] = w[(kstep*32+kmap(e))*ld + tn*16+l15]
// One thread per (tn, kstep, lane) chunk of 16 bf16 (32 bytes).
// ---------------------------------------------------------------------------
__global__ void wswizzle_kernel(const float* __restrict__ w, int ld, int ntiles,
                                unsigned short* __restrict__ bsw) {
    int idx = blockIdx.x * blockDim.x + threadIdx.x;   // (tn*16 + kstep)*32 + lane
    int lane = idx & 31;
    int kstep = (idx >> 5) & 15;
    int tn = idx >> 9;
    if (tn >= ntiles) return;
    int l15 = lane & 15, kgrp = lane >> 4;
    int col = tn * 16 + l15;
    unsigned short* dst = bsw + (size_t)idx * 16;
    #pragma unroll
    for (int e = 0; e < 16; ++e) {
        int k = kstep * 32 + kmap(e, kgrp);
        dst[e] = f2us(w[(size_t)k * ld + col]);
    }
}

// ---------------------------------------------------------------------------
// Kernel 4: x -> bf16 with Hilbert gather fused: Xp[b][np][:] = bf16(x[b][perm[np]][:])
// ---------------------------------------------------------------------------
__global__ void xconv_kernel(const float* __restrict__ x, const int* __restrict__ perm,
                             unsigned short* __restrict__ xp) {
    int tid = blockIdx.x * blockDim.x + threadIdx.x;   // one per 4 elements
    int c4 = tid & 127;                                // 512/4
    int np = (tid >> 7) & (NTOK - 1);
    int b = tid >> 20;
    const float4 v = *(const float4*)(x + ((size_t)b * NTOK + perm[np]) * CDIM + c4 * 4);
    unsigned r0 = (unsigned)f2us(v.x) | ((unsigned)f2us(v.y) << 16);
    unsigned r1 = (unsigned)f2us(v.z) | ((unsigned)f2us(v.w) << 16);
    uint2 pk = make_uint2(r0, r1);
    *(uint2*)(xp + ((size_t)b * NTOK + np) * CDIM + c4 * 4) = pk;
}

// ---------------------------------------------------------------------------
// Kernel 5: QKV GEMM (M=16384, N=1536, K=512) on bf16 operands.
// A = Xp (already permuted/converted), B = pre-swizzled qkv_w.
// Epilogue: +bias, Q scaled 1/8; Q,K as (b,s,h,128,64); V transposed (b,s,h,64,128).
// ---------------------------------------------------------------------------
__global__ __launch_bounds__(256) void qkv_gemm_kernel(
    const unsigned short* __restrict__ xp, const unsigned short* __restrict__ bsw,
    const float* __restrict__ bias,
    unsigned short* __restrict__ Qp, unsigned short* __restrict__ Kp,
    unsigned short* __restrict__ Vt) {
    const int wave = threadIdx.x >> 5, lane = threadIdx.x & 31;
    const int tile = blockIdx.x * 8 + wave;
    const int tn = tile % (C3 / 16);       // 0..95
    const int tm = tile / (C3 / 16);       // 0..1023
    const int l15 = lane & 15, kgrp = lane >> 4;

    const unsigned short* arow = xp + (size_t)(tm * 16 + l15) * CDIM;
    const unsigned short* bptr = bsw + (((size_t)tn * 16) * 32 + lane) * 16;

    v8f acc = {};
    for (int k0 = 0; k0 < CDIM; k0 += 32) {
        v16bf a = load_op(arow, k0, kgrp);
        v16bf bm = *(const v16bf*)bptr;
        bptr += 32 * 16;
        acc = __builtin_amdgcn_wmma_f32_16x16x32_bf16(
            false, a, false, bm, (short)0, acc, false, false);
    }

    const int ncol = tn * 16 + l15;
    const float bv = bias[ncol];
    const int t = ncol >> 9;                        // 0=q 1=k 2=v
    const int within = ncol & (CDIM - 1);
    const int h = within >> 6, dd = within & (HD - 1);
    const float scale = (t == 0) ? 0.125f : 1.0f;   // fold 1/sqrt(64) into Q
    #pragma unroll
    for (int r = 0; r < 8; ++r) {
        int row = tm * 16 + r + 8 * kgrp;           // permuted global row
        int bb = row >> 13, np = row & (NTOK - 1);
        int s = np >> 7, rin = np & (SEGL - 1);
        size_t base = (((size_t)bb * NSEG + s) * NH + h) * (SEGL * HD);
        unsigned short val = f2us((acc[r] + bv) * scale);
        if (t == 0)      Qp[base + (size_t)rin * HD + dd] = val;
        else if (t == 1) Kp[base + (size_t)rin * HD + dd] = val;
        else             Vt[base + (size_t)dd * SEGL + rin] = val;  // transposed
    }
}

// ---------------------------------------------------------------------------
// Kernel 6: segmented attention; one workgroup (8 waves) per (b,s,h).
// Wave w: 16x128 scores (16 WMMA), wave32 shfl softmax, P->LDS (layout
// redistribution), O = P@V (16 WMMA, V pre-transposed), scatter through perm.
// ---------------------------------------------------------------------------
__global__ __launch_bounds__(256) void seg_attn_kernel(
    const unsigned short* __restrict__ Qp, const unsigned short* __restrict__ Kp,
    const unsigned short* __restrict__ Vt, const int* __restrict__ perm,
    unsigned short* __restrict__ AO) {
    __shared__ unsigned short Pl[SEGL * SEGL];   // 32 KB of 320 KB WGP LDS

    const int bid = blockIdx.x;
    const int h = bid & (NH - 1);
    const int s = (bid >> 3) & (NSEG - 1);
    const int b = bid >> 9;
    const size_t base = (((size_t)b * NSEG + s) * NH + h) * (SEGL * HD);
    const unsigned short* Q = Qp + base;
    const unsigned short* K = Kp + base;
    const unsigned short* V = Vt + base;         // (64 x 128) transposed

    const int wave = threadIdx.x >> 5, lane = threadIdx.x & 31;
    const int l15 = lane & 15, kgrp = lane >> 4;

    // ---- scores S = (Q/8) K^T : rows [16w,16w+16), cols 0..127 ----
    v8f sc[8];
    const v8f vz = {};
    #pragma unroll
    for (int t = 0; t < 8; ++t) sc[t] = vz;
    const int am = wave * 16 + l15;
    const unsigned short* qrow = Q + am * HD;
    for (int k0 = 0; k0 < HD; k0 += 32) {
        v16bf a = load_op(qrow, k0, kgrp);
        #pragma unroll
        for (int t = 0; t < 8; ++t) {
            v16bf bm = load_op(K + (t * 16 + l15) * HD, k0, kgrp);
            sc[t] = __builtin_amdgcn_wmma_f32_16x16x32_bf16(
                false, a, false, bm, (short)0, sc[t], false, false);
        }
    }

    // ---- softmax per row; VGPR r holds row 16w + r + 8*kgrp, col t*16+l15 ----
    #pragma unroll
    for (int r = 0; r < 8; ++r) {
        float mx = -3.0e38f;
        #pragma unroll
        for (int t = 0; t < 8; ++t) mx = fmaxf(mx, sc[t][r]);
        #pragma unroll
        for (int off = 1; off < 16; off <<= 1)
            mx = fmaxf(mx, __shfl_xor(mx, off, 32));   // within 16-lane row group
        float ex[8];
        float sum = 0.f;
        #pragma unroll
        for (int t = 0; t < 8; ++t) { ex[t] = __expf(sc[t][r] - mx); sum += ex[t]; }
        #pragma unroll
        for (int off = 1; off < 16; off <<= 1)
            sum += __shfl_xor(sum, off, 32);
        const float inv = 1.0f / sum;
        const int row = wave * 16 + r + 8 * kgrp;
        #pragma unroll
        for (int t = 0; t < 8; ++t)
            Pl[row * SEGL + t * 16 + l15] = f2us(ex[t] * inv);
    }
    __syncthreads();

    // ---- O = P @ V : rows [16w,16w+16), N=64 (4 tiles), K=128 (4 steps) ----
    v8f oc[4];
    #pragma unroll
    for (int t = 0; t < 4; ++t) oc[t] = vz;
    const unsigned short* prow = &Pl[am * SEGL];
    for (int k0 = 0; k0 < SEGL; k0 += 32) {
        v16bf a = load_op(prow, k0, kgrp);            // ds_load_b128 pairs
        #pragma unroll
        for (int t = 0; t < 4; ++t) {
            v16bf bm = load_op(V + (t * 16 + l15) * SEGL, k0, kgrp);
            oc[t] = __builtin_amdgcn_wmma_f32_16x16x32_bf16(
                false, a, false, bm, (short)0, oc[t], false, false);
        }
    }

    // ---- scatter to original token order (fused inverse permutation) ----
    #pragma unroll
    for (int r = 0; r < 8; ++r) {
        const int row = wave * 16 + r + 8 * kgrp;
        const int tok = perm[s * SEGL + row];
        const size_t obase = ((size_t)b * NTOK + tok) * CDIM + h * HD;
        #pragma unroll
        for (int t = 0; t < 4; ++t)
            AO[obase + t * 16 + l15] = f2us(oc[t][r]);
    }
}

// ---------------------------------------------------------------------------
// Kernel 7: output GEMM (M=16384, N=512, K=512): bf16 AO x swizzled bf16 out_w.
// ---------------------------------------------------------------------------
__global__ __launch_bounds__(256) void out_gemm_kernel(
    const unsigned short* __restrict__ AO, const unsigned short* __restrict__ bsw,
    const float* __restrict__ bias, float* __restrict__ out) {
    const int wave = threadIdx.x >> 5, lane = threadIdx.x & 31;
    const int tile = blockIdx.x * 8 + wave;
    const int tn = tile & 31;              // 512/16
    const int tm = tile >> 5;              // 0..1023
    const int l15 = lane & 15, kgrp = lane >> 4;

    const unsigned short* arow = AO + (size_t)(tm * 16 + l15) * CDIM;
    const unsigned short* bptr = bsw + (((size_t)tn * 16) * 32 + lane) * 16;

    v8f acc = {};
    for (int k0 = 0; k0 < CDIM; k0 += 32) {
        v16bf a = load_op(arow, k0, kgrp);
        v16bf bm = *(const v16bf*)bptr;
        bptr += 32 * 16;
        acc = __builtin_amdgcn_wmma_f32_16x16x32_bf16(
            false, a, false, bm, (short)0, acc, false, false);
    }
    const int ncol = tn * 16 + l15;
    const float bv = bias[ncol];
    #pragma unroll
    for (int r = 0; r < 8; ++r) {
        const int row = tm * 16 + r + 8 * kgrp;
        out[(size_t)row * CDIM + ncol] = acc[r] + bv;
    }
}

// ---------------------------------------------------------------------------
// Host launcher
// ---------------------------------------------------------------------------
extern "C" void kernel_launch(void* const* d_in, const int* in_sizes, int n_in,
                              void* d_out, int out_size, void* d_ws, size_t ws_size,
                              hipStream_t stream) {
    const float* x      = (const float*)d_in[0];  // (2, 8192, 512)
    const float* qkv_w  = (const float*)d_in[1];  // (512, 1536)
    const float* qkv_b  = (const float*)d_in[2];  // (1536,)
    const float* out_w  = (const float*)d_in[3];  // (512, 512)
    const float* out_b  = (const float*)d_in[4];  // (512,)
    float* out          = (float*)d_out;          // (2, 8192, 512) fp32

    char* ws = (char*)d_ws;
    int* dvals = (int*)ws;                               // 32 KB
    int* perm  = (int*)(ws + (32 << 10));                // 32 KB
    const size_t QKV_ELEMS = (size_t)NB * NTOK * CDIM;   // 8,388,608 bf16 each
    unsigned short* Qp  = (unsigned short*)(ws + (128 << 10));
    unsigned short* Kp  = Qp + QKV_ELEMS;
    unsigned short* Vt  = Kp + QKV_ELEMS;
    unsigned short* AO  = Vt + QKV_ELEMS;                // bf16 (2,8192,512)
    unsigned short* Xp  = AO + QKV_ELEMS;                // bf16 (2,8192,512) permuted x
    unsigned short* Bq  = Xp + QKV_ELEMS;                // swizzled qkv_w (1.5 MB)
    unsigned short* Bo  = Bq + (size_t)96 * 16 * 32 * 16;// swizzled out_w (0.5 MB)

    // 1) Hilbert permutation
    hilbert_d_kernel<<<NTOK / 256, 256, 0, stream>>>(dvals);
    hilbert_rank_kernel<<<NTOK / 256, 256, 0, stream>>>(dvals, perm);

    // 2) one-shot operand prep: swizzled bf16 weights + permuted bf16 x
    wswizzle_kernel<<<(96 * 16 * 32) / 256, 256, 0, stream>>>(qkv_w, C3, 96, Bq);
    wswizzle_kernel<<<(32 * 16 * 32) / 256, 256, 0, stream>>>(out_w, CDIM, 32, Bo);
    xconv_kernel<<<(int)(QKV_ELEMS / 4 / 256), 256, 0, stream>>>(x, perm, Xp);

    // 3) QKV GEMM -> Q,K tiles + transposed V tiles (bf16)
    {
        const int tiles = ((NB * NTOK) / 16) * (C3 / 16);   // 98304
        qkv_gemm_kernel<<<tiles / 8, 256, 0, stream>>>(Xp, Bq, qkv_b, Qp, Kp, Vt);
    }

    // 4) segmented attention, scattered back to original token order
    seg_attn_kernel<<<NB * NSEG * NH, 256, 0, stream>>>(Qp, Kp, Vt, perm, AO);

    // 5) output projection -> fp32 d_out
    {
        const int tiles = ((NB * NTOK) / 16) * (CDIM / 16); // 32768
        out_gemm_kernel<<<tiles / 8, 256, 0, stream>>>(AO, Bo, out_b, out);
    }
}
